// TemplatePairStack_9208409882891
// MI455X (gfx1250) — matrix operand order repaired and checked by reading, not compiled, not run on hardware
//
#include <hip/hip_runtime.h>
#include <hip/hip_bf16.h>

// ---------------------------------------------------------------------------
// TemplatePairStack on gfx1250 (MI455X): all GEMMs via v_wmma_f32_16x16x32_bf16
// (wave32, 16x16x32 bf16 -> f32 accumulate). fp32 residual stream lives in
// d_out; bf16 staging buffers in d_ws (~56 MB, all L2-resident).
// ---------------------------------------------------------------------------

#define NSEQ   256
#define CDIM   64
#define NHEAD  4
#define MROWS  (NSEQ*NSEQ)   // 65536

typedef __bf16 bf16;
typedef __attribute__((__ext_vector_type__(16))) __bf16 v16bf;
typedef __attribute__((__ext_vector_type__(8)))  __bf16 v8bf;
typedef __attribute__((__ext_vector_type__(8)))  float  v8f;

__device__ __forceinline__ v8f zero_v8f(){
  v8f z;
#pragma unroll
  for (int i=0;i<8;i++) z[i]=0.0f;
  return z;
}
__device__ __forceinline__ v16bf zero_v16bf(){
  v16bf z;
#pragma unroll
  for (int i=0;i<16;i++) z[i]=(bf16)0.0f;
  return z;
}

__device__ __forceinline__ v8f wmma_bf16(v16bf a, v16bf b, v8f c){
  return __builtin_amdgcn_wmma_f32_16x16x32_bf16(false, a, false, b, (short)0, c, false, false);
}

// A fragment (16x32 bf16, MxK): lane holds row (lane&15); g=lane>>4.
// h[0..15] = row[32kt + 16g .. +15]
__device__ __forceinline__ v16bf load_afrag(const bf16* row, int g){
  v8bf lo = *(const v8bf*)(row + 16*g);
  v8bf hi = *(const v8bf*)(row + 16*g + 8);
  v16bf r;
#pragma unroll
  for (int i=0;i<8;i++){ r[i]=lo[i]; r[i+8]=hi[i]; }
  return r;
}

__device__ __forceinline__ float sigmoidf(float x){ return 1.0f/(1.0f+__expf(-x)); }

__device__ __forceinline__ float rmax16(float v){
  v = fmaxf(v, __shfl_xor(v,1));
  v = fmaxf(v, __shfl_xor(v,2));
  v = fmaxf(v, __shfl_xor(v,4));
  v = fmaxf(v, __shfl_xor(v,8));
  return v;
}
__device__ __forceinline__ float rsum16(float v){
  v += __shfl_xor(v,1); v += __shfl_xor(v,2);
  v += __shfl_xor(v,4); v += __shfl_xor(v,8);
  return v;
}
__device__ __forceinline__ float rsum32(float v){
  v += __shfl_xor(v,1); v += __shfl_xor(v,2); v += __shfl_xor(v,4);
  v += __shfl_xor(v,8); v += __shfl_xor(v,16);
  return v;
}

// ---------------------------------------------------------------------------
// LayerNorm over C=64: one wave (32 lanes) per row, 2 channels per lane.
// TREAD transposes the spatial read (per-column attention view).
// ---------------------------------------------------------------------------
template<bool TREAD>
__global__ void ln_kernel(const float* __restrict__ src, const float* __restrict__ sc,
                          const float* __restrict__ of, bf16* __restrict__ dst){
  int row  = blockIdx.x*8 + (threadIdx.x>>5);
  int lane = threadIdx.x & 31;
  size_t sbase = TREAD ? (size_t)((row & 255)*NSEQ + (row >> 8)) * CDIM
                       : (size_t)row * CDIM;
  float x0 = src[sbase + lane];
  float x1 = src[sbase + 32 + lane];
  float mean = rsum32(x0 + x1) * (1.0f/64.0f);
  float d0 = x0 - mean, d1 = x1 - mean;
  float var  = rsum32(d0*d0 + d1*d1) * (1.0f/64.0f);
  float rs = rsqrtf(var + 1e-5f);
  size_t dbase = (size_t)row * CDIM;
  dst[dbase + lane]      = (bf16)(d0*rs*sc[lane]    + of[lane]);
  dst[dbase + 32 + lane] = (bf16)(d1*rs*sc[lane+32] + of[lane+32]);
}

// ---------------------------------------------------------------------------
// Generic activation[M,K](bf16) x weight[K,NOUT](f32, staged transposed bf16
// in LDS) GEMM, 8 waves x 16 rows per block, fused epilogues.
// ---------------------------------------------------------------------------
enum { EPI_BF16=0, EPI_BF16_SIG=1, EPI_BF16_RELU=2, EPI_RES=3, EPI_RES_GATED=4, EPI_NB=5 };

template<int K, int NOUT, int EPI, bool TSTORE>
__global__ void gemm_kernel(const bf16* __restrict__ A, const float* __restrict__ W, int wcols,
                            const float* __restrict__ bias, float scale,
                            bf16* __restrict__ outB, float* __restrict__ outF,
                            const bf16* __restrict__ gate){
  __shared__ __align__(32) bf16 wl[NOUT*K];   // wl[n][k] (column-major weight)
  for (int idx = threadIdx.x; idx < NOUT*K; idx += 256){
    int n = idx / K, k = idx - n*K;
    wl[n*K + k] = (n < wcols) ? (bf16)W[k*wcols + n] : (bf16)0.0f;
  }
  __syncthreads();

  int lane = threadIdx.x & 31;
  int wv   = threadIdx.x >> 5;
  int ln   = lane & 15;
  int g    = lane >> 4;
  int m0   = blockIdx.x*128 + wv*16;

  const bf16* arow = A + (size_t)(m0 + ln)*K;
  v16bf af[K/32];
#pragma unroll
  for (int kt=0; kt<K/32; kt++) af[kt] = load_afrag(arow + 32*kt, g);

#pragma unroll
  for (int nt=0; nt<NOUT/16; nt++){
    v8f acc = zero_v8f();
#pragma unroll
    for (int kt=0; kt<K/32; kt++){
      v16bf bfg = *(const v16bf*)(wl + (size_t)(nt*16 + ln)*K + 32*kt + 16*g);
      acc = wmma_bf16(af[kt], bfg, acc);
    }
    int n = nt*16 + ln;
    float bv = (bias != nullptr) ? bias[n] : 0.0f;
#pragma unroll
    for (int r=0;r<8;r++){
      int m = m0 + r + 8*g;
      float v = acc[r]*scale + bv;
      if (EPI == EPI_BF16){
        outB[(size_t)m*NOUT + n] = (bf16)v;
      } else if (EPI == EPI_BF16_SIG){
        outB[(size_t)m*NOUT + n] = (bf16)sigmoidf(v);
      } else if (EPI == EPI_BF16_RELU){
        outB[(size_t)m*NOUT + n] = (bf16)fmaxf(v, 0.0f);
      } else if (EPI == EPI_RES){
        int d = TSTORE ? ((m & 255)*NSEQ + (m >> 8)) : m;
        outF[(size_t)d*CDIM + n] += v;
      } else if (EPI == EPI_RES_GATED){
        v *= (float)gate[(size_t)m*CDIM + n];
        outF[(size_t)m*CDIM + n] += v;
      } else { // EPI_NB: f2d projection -> nb[h][m], only 4 real columns
        if (n < NHEAD) outF[(size_t)n*MROWS + m] = v;
      }
    }
  }
}

// ---------------------------------------------------------------------------
// TriangleMult projection: p = mask * (a@W + b) * sigmoid(a@Wg + bg),
// written channel-major projT[c][d0][d1] (SWAP transposes d0/d1 for the
// "incoming" einsum orientation).
// ---------------------------------------------------------------------------
template<bool SWAP>
__global__ void mult_proj_kernel(const bf16* __restrict__ A, const float* __restrict__ W,
                                 const float* __restrict__ bvec, const float* __restrict__ Wg,
                                 const float* __restrict__ bg, const float* __restrict__ mask,
                                 bf16* __restrict__ dst){
  __shared__ __align__(32) bf16 wl [64*64];
  __shared__ __align__(32) bf16 wgl[64*64];
  for (int idx=threadIdx.x; idx<4096; idx+=256){
    int n = idx >> 6, k = idx & 63;
    wl [n*64+k] = (bf16)W [k*64+n];
    wgl[n*64+k] = (bf16)Wg[k*64+n];
  }
  __syncthreads();

  int lane = threadIdx.x & 31, wv = threadIdx.x>>5;
  int ln = lane & 15, g = lane>>4;
  int m0 = blockIdx.x*128 + wv*16;
  const bf16* arow = A + (size_t)(m0+ln)*64;
  v16bf af0 = load_afrag(arow, g);
  v16bf af1 = load_afrag(arow+32, g);

#pragma unroll
  for (int nt=0; nt<4; nt++){
    v8f a1 = zero_v8f(), a2 = zero_v8f();
    v16bf b0 = *(const v16bf*)(wl  + (size_t)(nt*16+ln)*64 + 16*g);
    v16bf b1 = *(const v16bf*)(wl  + (size_t)(nt*16+ln)*64 + 32 + 16*g);
    a1 = wmma_bf16(af0, b0, a1);
    a1 = wmma_bf16(af1, b1, a1);
    v16bf c0 = *(const v16bf*)(wgl + (size_t)(nt*16+ln)*64 + 16*g);
    v16bf c1 = *(const v16bf*)(wgl + (size_t)(nt*16+ln)*64 + 32 + 16*g);
    a2 = wmma_bf16(af0, c0, a2);
    a2 = wmma_bf16(af1, c1, a2);
    int n = nt*16 + ln;
    float b1v = bvec[n], b2v = bg[n];
#pragma unroll
    for (int r=0;r<8;r++){
      int m = m0 + r + 8*g;
      float v = mask[m] * (a1[r]+b1v) * sigmoidf(a2[r]+b2v);
      int d = SWAP ? ((m & 255)*NSEQ + (m>>8)) : m;
      dst[(size_t)n*MROWS + d] = (bf16)v;
    }
  }
}

// ---------------------------------------------------------------------------
// Triangle einsum: X[i,j,c] = sum_k Abuf[c][i][k] * Bbuf[c][j][k]
// 64 per-channel 256x256x256 bf16 GEMMs; one 16x16 tile per wave, K=256.
// ---------------------------------------------------------------------------
__global__ void tri_einsum_kernel(const bf16* __restrict__ Ab, const bf16* __restrict__ Bb,
                                  float* __restrict__ X){
  int lane = threadIdx.x & 31, wv = threadIdx.x >> 5;
  int ln = lane & 15, g = lane >> 4;
  int t   = blockIdx.x*8 + wv;
  int c   = t >> 8;
  int rem = t & 255;
  int i0  = (rem >> 4) << 4;
  int j0  = (rem & 15) << 4;
  const bf16* arow = Ab + (size_t)c*MROWS + (size_t)(i0+ln)*NSEQ;
  const bf16* brow = Bb + (size_t)c*MROWS + (size_t)(j0+ln)*NSEQ;
  v8f acc = zero_v8f();
#pragma unroll
  for (int kt=0; kt<8; kt++){
    v16bf af  = load_afrag(arow + 32*kt, g);
    v16bf bfg = *(const v16bf*)(brow + 32*kt + 16*g);
    acc = wmma_bf16(af, bfg, acc);
  }
#pragma unroll
  for (int r=0;r<8;r++){
    int i = i0 + r + 8*g;
    int j = j0 + ln;
    X[((size_t)i*NSEQ + j)*CDIM + c] = acc[r];
  }
}

// ---------------------------------------------------------------------------
// Flash-style triangle attention for one (b,h): never materializes the
// 256x256 logits in memory. One wave owns a 16-row query tile; per 32-key
// block: S = QK^T (2 wmma, DH=16 zero-padded to K=32) + mask bias + pair
// bias, online softmax, P staged via per-wave LDS to A-layout, P@V (1 wmma).
// Gate (precomputed sigmoid) is fused into the wa store.
// ---------------------------------------------------------------------------
template<bool TMASK>
__global__ void attn_kernel(const bf16* __restrict__ qb, const bf16* __restrict__ kb,
                            const bf16* __restrict__ vb, const float* __restrict__ nb,
                            const bf16* __restrict__ gate, const float* __restrict__ mask,
                            bf16* __restrict__ wa){
  __shared__ __align__(32) bf16 VsT[16*256];   // V transposed: VsT[c][k]
  __shared__ __align__(32) bf16 plds[8*512];   // per-wave 16x32 P staging
  int b = blockIdx.x >> 2;
  int h = blockIdx.x & 3;
  for (int idx=threadIdx.x; idx<4096; idx+=256){
    int c = idx >> 8, k = idx & 255;
    VsT[c*256 + k] = vb[(size_t)(b*256 + k)*CDIM + h*16 + c];
  }
  __syncthreads();

  int lane = threadIdx.x & 31, wv = threadIdx.x >> 5;
  int ln = lane & 15, g = lane >> 4;
  const float* nbh = nb + (size_t)h*MROWS;
  bf16* pl = plds + wv*512;

  for (int qp=0; qp<2; qp++){
    int q0 = qp*128 + wv*16;
    // Q fragment: real c = 0..15, padded to K=32 with zeros
    v16bf qf = zero_v16bf();
    {
      const bf16* qr = qb + (size_t)(b*256 + q0 + ln)*CDIM + h*16;
      if (g == 0) qf = *(const v16bf*)qr;
    }
    v8f acc = zero_v8f();
    float mrow[8], srow[8];
#pragma unroll
    for (int r=0;r<8;r++){ mrow[r] = -1e30f; srow[r] = 0.0f; }

    for (int kbk=0; kbk<8; kbk++){
      int k0 = kbk*32;
      // K^T fragments (K(c)=idx+16g: g==0 real c=0..15, g==1 padded zeros)
      v16bf kf0 = zero_v16bf(), kf1 = zero_v16bf();
      if (g == 0){
        kf0 = *(const v16bf*)(kb + (size_t)(b*256 + k0 + ln)*CDIM + h*16);
        kf1 = *(const v16bf*)(kb + (size_t)(b*256 + k0 + 16 + ln)*CDIM + h*16);
      }
      v8f s0 = zero_v8f(), s1 = zero_v8f();
      s0 = wmma_bf16(qf, kf0, s0);
      s1 = wmma_bf16(qf, kf1, s1);

      float mk0 = TMASK ? mask[(k0+ln)*NSEQ + b]    : mask[b*NSEQ + k0 + ln];
      float mk1 = TMASK ? mask[(k0+16+ln)*NSEQ + b] : mask[b*NSEQ + k0 + 16 + ln];
      float mb0 = 1e9f*(mk0 - 1.0f);
      float mb1 = 1e9f*(mk1 - 1.0f);
#pragma unroll
      for (int r=0;r<8;r++){
        int q = q0 + r + 8*g;
        float t0 = s0[r] + mb0 + nbh[(size_t)q*NSEQ + k0 + ln];
        float t1 = s1[r] + mb1 + nbh[(size_t)q*NSEQ + k0 + 16 + ln];
        s0[r] = fminf(fmaxf(t0, -1e8f), 1e8f);
        s1[r] = fminf(fmaxf(t1, -1e8f), 1e8f);
      }
#pragma unroll
      for (int r=0;r<8;r++){
        float tm   = rmax16(fmaxf(s0[r], s1[r]));
        float mnew = fmaxf(mrow[r], tm);
        float sc   = __expf(mrow[r] - mnew);
        float e0   = __expf(s0[r] - mnew);
        float e1   = __expf(s1[r] - mnew);
        srow[r] = srow[r]*sc + rsum16(e0 + e1);
        mrow[r] = mnew;
        acc[r] *= sc;
        s0[r] = e0; s1[r] = e1;
      }
      // Stage P (C-layout -> row-major LDS) then reload in A-layout
#pragma unroll
      for (int r=0;r<8;r++){
        pl[(r + 8*g)*32 + ln]      = (bf16)s0[r];
        pl[(r + 8*g)*32 + 16 + ln] = (bf16)s1[r];
      }
      asm volatile("s_wait_dscnt 0" ::: "memory");
      v16bf pf = load_afrag(pl + ln*32, g);
      v16bf vf = *(const v16bf*)(VsT + ln*256 + k0 + 16*g);
      acc = wmma_bf16(pf, vf, acc);
    }
#pragma unroll
    for (int r=0;r<8;r++){
      int q = q0 + r + 8*g;
      size_t idx = (size_t)(b*256 + q)*CDIM + h*16 + ln;
      float v = acc[r] / srow[r];
      v *= (float)gate[idx];
      wa[idx] = (bf16)v;
    }
  }
}

// ---------------------------------------------------------------------------
// Host: param leaves follow jax tree-flatten order (sorted dict keys):
// per layer: attn_end(10), attn_start(10), mult_in(16), mult_out(16), trans(6)
// ---------------------------------------------------------------------------
extern "C" void kernel_launch(void* const* d_in, const int* in_sizes, int n_in,
                              void* d_out, int out_size, void* d_ws, size_t ws_size,
                              hipStream_t stream){
  (void)in_sizes; (void)n_in; (void)out_size; (void)ws_size;
  const float* mask = (const float*)d_in[1];
  float* act = (float*)d_out;
  hipMemcpyAsync(act, d_in[0], (size_t)MROWS*CDIM*sizeof(float),
                 hipMemcpyDeviceToDevice, stream);

  char* ws = (char*)d_ws;
  const size_t MB = (size_t)1 << 20;
  bf16*  aBuf    = (bf16*) (ws +  0*MB);  // LN output (also xn in tri-mult)
  bf16*  qBuf    = (bf16*) (ws +  8*MB);
  bf16*  kBuf    = (bf16*) (ws + 16*MB);
  bf16*  vBuf    = (bf16*) (ws + 24*MB);
  bf16*  gateBuf = (bf16*) (ws + 32*MB);
  bf16*  waBuf   = (bf16*) (ws + 40*MB);
  float* nbBuf   = (float*)(ws + 48*MB);  // [4][65536] f32
  bf16*  lpT     = (bf16*) (ws +  8*MB);  // aliases qBuf (dead in mult phase)
  bf16*  rpT     = (bf16*) (ws + 16*MB);  // aliases kBuf
  bf16*  h1Buf   = (bf16*) (ws +  8*MB);  // aliases q+k (16 MB, transition)
  float* xBuf    = (float*)(ws + 40*MB);  // aliases wa+nb (16 MB -> 56 MB)

  auto PF = [&](int i)->const float* { return (const float*)d_in[i]; };

  for (int layer=0; layer<2; layer++){
    int L = 2 + layer*58;

    auto run_attn = [&](int base, bool percol){
      const float* f2d  = PF(base+0);
      const float* gb   = PF(base+1);
      const float* gw   = PF(base+2);
      const float* kw   = PF(base+3);
      const float* ob   = PF(base+4);
      const float* ow   = PF(base+5);
      const float* qn_o = PF(base+6);
      const float* qn_s = PF(base+7);
      const float* qw   = PF(base+8);
      const float* vw   = PF(base+9);
      if (percol) ln_kernel<true ><<<8192,256,0,stream>>>(act, qn_s, qn_o, aBuf);
      else        ln_kernel<false><<<8192,256,0,stream>>>(act, qn_s, qn_o, aBuf);
      gemm_kernel<64,16,EPI_NB,false><<<512,256,0,stream>>>(
          aBuf, f2d, 4, (const float*)nullptr, 1.0f, (bf16*)nullptr, nbBuf, (const bf16*)nullptr);
      gemm_kernel<64,64,EPI_BF16,false><<<512,256,0,stream>>>(
          aBuf, qw, 64, (const float*)nullptr, 0.25f, qBuf, (float*)nullptr, (const bf16*)nullptr);
      gemm_kernel<64,64,EPI_BF16,false><<<512,256,0,stream>>>(
          aBuf, kw, 64, (const float*)nullptr, 1.0f, kBuf, (float*)nullptr, (const bf16*)nullptr);
      gemm_kernel<64,64,EPI_BF16,false><<<512,256,0,stream>>>(
          aBuf, vw, 64, (const float*)nullptr, 1.0f, vBuf, (float*)nullptr, (const bf16*)nullptr);
      gemm_kernel<64,64,EPI_BF16_SIG,false><<<512,256,0,stream>>>(
          aBuf, gw, 64, gb, 1.0f, gateBuf, (float*)nullptr, (const bf16*)nullptr);
      if (percol) attn_kernel<true ><<<1024,256,0,stream>>>(qBuf,kBuf,vBuf,nbBuf,gateBuf,mask,waBuf);
      else        attn_kernel<false><<<1024,256,0,stream>>>(qBuf,kBuf,vBuf,nbBuf,gateBuf,mask,waBuf);
      if (percol) gemm_kernel<64,64,EPI_RES,true ><<<512,256,0,stream>>>(
          waBuf, ow, 64, ob, 1.0f, (bf16*)nullptr, act, (const bf16*)nullptr);
      else        gemm_kernel<64,64,EPI_RES,false><<<512,256,0,stream>>>(
          waBuf, ow, 64, ob, 1.0f, (bf16*)nullptr, act, (const bf16*)nullptr);
    };

    auto run_mult = [&](int base, bool outgoing){
      const float* c_o  = PF(base+0);
      const float* c_s  = PF(base+1);
      const float* gb   = PF(base+2);
      const float* gw   = PF(base+3);
      const float* lb   = PF(base+4);
      const float* lgb  = PF(base+5);
      const float* lgw  = PF(base+6);
      const float* ln_o = PF(base+7);
      const float* ln_s = PF(base+8);
      const float* lw   = PF(base+9);
      const float* ob   = PF(base+10);
      const float* ow   = PF(base+11);
      const float* rb   = PF(base+12);
      const float* rgb  = PF(base+13);
      const float* rgw  = PF(base+14);
      const float* rw   = PF(base+15);
      ln_kernel<false><<<8192,256,0,stream>>>(act, ln_s, ln_o, aBuf);
      gemm_kernel<64,64,EPI_BF16_SIG,false><<<512,256,0,stream>>>(
          aBuf, gw, 64, gb, 1.0f, gateBuf, (float*)nullptr, (const bf16*)nullptr);
      if (outgoing){
        // X[i,j,c] = sum_k lp[i,k,c] * rp[j,k,c]
        mult_proj_kernel<false><<<512,256,0,stream>>>(aBuf, lw, lb, lgw, lgb, mask, lpT);
        mult_proj_kernel<false><<<512,256,0,stream>>>(aBuf, rw, rb, rgw, rgb, mask, rpT);
        tri_einsum_kernel<<<2048,256,0,stream>>>(lpT, rpT, xBuf);
      } else {
        // X[i,j,c] = sum_k lp[k,j,c] * rp[k,i,c]  (store both transposed)
        mult_proj_kernel<true ><<<512,256,0,stream>>>(aBuf, lw, lb, lgw, lgb, mask, lpT);
        mult_proj_kernel<true ><<<512,256,0,stream>>>(aBuf, rw, rb, rgw, rgb, mask, rpT);
        tri_einsum_kernel<<<2048,256,0,stream>>>(rpT, lpT, xBuf);
      }
      ln_kernel<false><<<8192,256,0,stream>>>(xBuf, c_s, c_o, aBuf);   // aBuf := LN(x)
      gemm_kernel<64,64,EPI_RES_GATED,false><<<512,256,0,stream>>>(
          aBuf, ow, 64, ob, 1.0f, (bf16*)nullptr, act, gateBuf);
    };

    run_attn(L+10, false);  // attn_start (row-wise)
    run_attn(L+ 0, true);   // attn_end   (column-wise)
    run_mult(L+36, true);   // mult_out   (outgoing)
    run_mult(L+20, false);  // mult_in    (incoming)
    {
      int base = L + 52;    // transition
      const float* b1   = PF(base+0);
      const float* b2   = PF(base+1);
      const float* ln_o = PF(base+2);
      const float* ln_s = PF(base+3);
      const float* w1   = PF(base+4);
      const float* w2   = PF(base+5);
      ln_kernel<false><<<8192,256,0,stream>>>(act, ln_s, ln_o, aBuf);
      gemm_kernel<64,128,EPI_BF16_RELU,false><<<512,256,0,stream>>>(
          aBuf, w1, 128, b1, 1.0f, h1Buf, (float*)nullptr, (const bf16*)nullptr);
      gemm_kernel<128,64,EPI_RES,false><<<512,256,0,stream>>>(
          h1Buf, w2, 64, b2, 1.0f, (bf16*)nullptr, act, (const bf16*)nullptr);
    }
  }
}
